// TransformerAttention_45947560133318
// MI455X (gfx1250) — matrix operand order repaired and checked by reading, not compiled, not run on hardware
//
#include <hip/hip_runtime.h>

#define D_MODEL 512
#define SEQ     2048
#define BATCH   4
#define NHEADS  8
#define DK      64
#define MTOT    (BATCH * SEQ)   // 8192

#define AS1 __attribute__((address_space(1)))
#define AS3 __attribute__((address_space(3)))

#if __has_builtin(__builtin_amdgcn_global_load_async_to_lds_b128) && \
    __has_builtin(__builtin_amdgcn_s_wait_asynccnt)
#define USE_ASYNC_LDS 1
#else
#define USE_ASYNC_LDS 0
#endif

typedef int v4i __attribute__((vector_size(16)));
typedef __attribute__((ext_vector_type(16))) __bf16 v16bf;
typedef __attribute__((ext_vector_type(8)))  float  v8f;

union FragB16 {
    v16bf v;
    uint4 q[2];
    unsigned short u[16];
};

__device__ __forceinline__ unsigned short f32_to_bf16(float f) {
    union { float f; unsigned int u; } c;
    c.f = f;
    unsigned int u = c.u;
    u += 0x7FFFu + ((u >> 16) & 1u);   // round-to-nearest-even
    return (unsigned short)(u >> 16);
}

#if USE_ASYNC_LDS
__device__ __forceinline__ void async_copy16(const unsigned short* g, unsigned short* l) {
    __builtin_amdgcn_global_load_async_to_lds_b128((AS1 v4i*)g, (AS3 v4i*)l, 0, 0);
}
#endif

// ---------------------------------------------------------------------------
// elementwise converts (vectorized: float4 -> 4x bf16 packed in 8B)
// ---------------------------------------------------------------------------
__global__ void cvt_f32_bf16_v4(const float* __restrict__ src,
                                unsigned short* __restrict__ dst, int n4) {
    int i = blockIdx.x * blockDim.x + threadIdx.x;
    if (i < n4) {
        float4 f = reinterpret_cast<const float4*>(src)[i];
        unsigned long long p =
              (unsigned long long)f32_to_bf16(f.x)
            | ((unsigned long long)f32_to_bf16(f.y) << 16)
            | ((unsigned long long)f32_to_bf16(f.z) << 32)
            | ((unsigned long long)f32_to_bf16(f.w) << 48);
        reinterpret_cast<unsigned long long*>(dst)[i] = p;
    }
}

__global__ void copy_f32(const float* __restrict__ src,
                         float* __restrict__ dst, int n) {
    int i = blockIdx.x * blockDim.x + threadIdx.x;
    if (i < n) dst[i] = src[i];
}

// ---------------------------------------------------------------------------
// Generic bf16 WMMA GEMM:  C[m,n] = sum_k A[m,k] * W[n,k]   (K = 512 fixed)
// Block tile 128x128, 8 waves, each wave = 32x64 (2x4 WMMA tiles).
// Double-buffered LDS fed by GLOBAL_LOAD_ASYNC_TO_LDS_B128 (ASYNCcnt).
// mode 0: QKV  (N=1536) -> Q bf16, K bf16, V transposed bf16 [B,H,dk,S]
// mode 1: O|G  (N=1024) -> two f32 buffers
// mode 2: MU|LV(N=1024) -> +bias, two f32 buffers
// ---------------------------------------------------------------------------
__global__ __launch_bounds__(256) void gemm_bf16(
    const unsigned short* __restrict__ A,
    const unsigned short* __restrict__ W,
    int mode,
    unsigned short* __restrict__ oq,
    unsigned short* __restrict__ okk,
    unsigned short* __restrict__ ovt,
    float* __restrict__ of0,
    float* __restrict__ of1,
    const float* __restrict__ bias)
{
    // 48-half row stride: 96B rows keep 16B alignment for ds_load_b128
    __shared__ unsigned short As[2][128 * 48];
    __shared__ unsigned short Bs[2][128 * 48];

    const int tid   = threadIdx.x;
    const int m0    = blockIdx.x * 128;
    const int n0    = blockIdx.y * 128;
    const int wave  = tid >> 5;
    const int lane  = tid & 31;
    const int wm    = wave >> 1;      // 0..3 -> 32 rows each
    const int wn    = wave & 1;       // 0..1 -> 64 cols each
    const int lhalf = lane >> 4;      // lane 0-15 / 16-31
    const int l15   = lane & 15;

    // per-thread staging chunks: chunk = tid + c*256 -> (row, kc)
    const int row0 = tid >> 2,           kc0 = (tid & 3) * 8;
    const int row1 = (tid + 256) >> 2,   kc1 = ((tid + 256) & 3) * 8;

    v8f acc[2][4];
#pragma unroll
    for (int mt = 0; mt < 2; ++mt)
#pragma unroll
        for (int nt = 0; nt < 4; ++nt)
            acc[mt][nt] = (v8f){0.f, 0.f, 0.f, 0.f, 0.f, 0.f, 0.f, 0.f};

#if USE_ASYNC_LDS
    // ---- async double-buffered pipeline ----
    auto issue = [&](int kb, int buf) {
        async_copy16(&A[(size_t)(m0 + row0) * D_MODEL + kb + kc0], &As[buf][row0 * 48 + kc0]);
        async_copy16(&A[(size_t)(m0 + row1) * D_MODEL + kb + kc1], &As[buf][row1 * 48 + kc1]);
        async_copy16(&W[(size_t)(n0 + row0) * D_MODEL + kb + kc0], &Bs[buf][row0 * 48 + kc0]);
        async_copy16(&W[(size_t)(n0 + row1) * D_MODEL + kb + kc1], &Bs[buf][row1 * 48 + kc1]);
    };
    issue(0, 0);
    int ib = 0;
    for (int kb = 0; kb < D_MODEL; kb += 32, ib ^= 1) {
        const bool hasnext = (kb + 32) < D_MODEL;
        if (hasnext) {
            issue(kb + 32, ib ^ 1);                 // prev iter's barrier makes buf free
            __builtin_amdgcn_s_wait_asynccnt(4);    // cur buffer landed; next still in flight
        } else {
            __builtin_amdgcn_s_wait_asynccnt(0);
        }
        __syncthreads();                            // cur buffer visible to all waves

        FragB16 af[2];
#pragma unroll
        for (int mt = 0; mt < 2; ++mt) {
            const unsigned short* ap = &As[ib][(wm * 32 + mt * 16 + l15) * 48 + lhalf * 8];
            af[mt].q[0] = *reinterpret_cast<const uint4*>(ap);
            af[mt].q[1] = *reinterpret_cast<const uint4*>(ap + 16);
        }
#pragma unroll
        for (int nt = 0; nt < 4; ++nt) {
            FragB16 bf;
            const unsigned short* bp = &Bs[ib][(wn * 64 + nt * 16 + l15) * 48 + lhalf * 8];
            bf.q[0] = *reinterpret_cast<const uint4*>(bp);
            bf.q[1] = *reinterpret_cast<const uint4*>(bp + 16);
#pragma unroll
            for (int mt = 0; mt < 2; ++mt)
                acc[mt][nt] = __builtin_amdgcn_wmma_f32_16x16x32_bf16(
                    false, af[mt].v, false, bf.v, (short)0, acc[mt][nt], false, false);
        }
        __syncthreads();                            // all waves done reading cur buffer
    }
#else
    // ---- fallback: synchronous staging ----
    for (int kb = 0; kb < D_MODEL; kb += 32) {
        *reinterpret_cast<uint4*>(&As[0][row0 * 48 + kc0]) =
            *reinterpret_cast<const uint4*>(&A[(size_t)(m0 + row0) * D_MODEL + kb + kc0]);
        *reinterpret_cast<uint4*>(&As[0][row1 * 48 + kc1]) =
            *reinterpret_cast<const uint4*>(&A[(size_t)(m0 + row1) * D_MODEL + kb + kc1]);
        *reinterpret_cast<uint4*>(&Bs[0][row0 * 48 + kc0]) =
            *reinterpret_cast<const uint4*>(&W[(size_t)(n0 + row0) * D_MODEL + kb + kc0]);
        *reinterpret_cast<uint4*>(&Bs[0][row1 * 48 + kc1]) =
            *reinterpret_cast<const uint4*>(&W[(size_t)(n0 + row1) * D_MODEL + kb + kc1]);
        __syncthreads();

        FragB16 af[2];
#pragma unroll
        for (int mt = 0; mt < 2; ++mt) {
            const unsigned short* ap = &As[0][(wm * 32 + mt * 16 + l15) * 48 + lhalf * 8];
            af[mt].q[0] = *reinterpret_cast<const uint4*>(ap);
            af[mt].q[1] = *reinterpret_cast<const uint4*>(ap + 16);
        }
#pragma unroll
        for (int nt = 0; nt < 4; ++nt) {
            FragB16 bf;
            const unsigned short* bp = &Bs[0][(wn * 64 + nt * 16 + l15) * 48 + lhalf * 8];
            bf.q[0] = *reinterpret_cast<const uint4*>(bp);
            bf.q[1] = *reinterpret_cast<const uint4*>(bp + 16);
#pragma unroll
            for (int mt = 0; mt < 2; ++mt)
                acc[mt][nt] = __builtin_amdgcn_wmma_f32_16x16x32_bf16(
                    false, af[mt].v, false, bf.v, (short)0, acc[mt][nt], false, false);
        }
        __syncthreads();
    }
#endif

    // epilogue: C layout -> VGPR r holds row (r | r+8 by lane half), col = l15
#pragma unroll
    for (int mt = 0; mt < 2; ++mt) {
#pragma unroll
        for (int nt = 0; nt < 4; ++nt) {
#pragma unroll
            for (int r = 0; r < 8; ++r) {
                int m = m0 + wm * 32 + mt * 16 + lhalf * 8 + r;
                int n = n0 + wn * 64 + nt * 16 + l15;
                float val = acc[mt][nt][r];
                if (mode == 0) {
                    int sec = n >> 9, nn = n & 511;
                    if (sec == 0) {
                        oq[(size_t)m * D_MODEL + nn] = f32_to_bf16(val);
                    } else if (sec == 1) {
                        okk[(size_t)m * D_MODEL + nn] = f32_to_bf16(val);
                    } else {
                        int bb = m >> 11, s = m & 2047, hh = nn >> 6, d = nn & 63;
                        ovt[(((size_t)((bb * NHEADS + hh) * DK + d)) << 11) + s] =
                            f32_to_bf16(val);
                    }
                } else if (mode == 1) {
                    if (n < 512) of0[(size_t)m * D_MODEL + n] = val;
                    else         of1[(size_t)m * D_MODEL + (n - 512)] = val;
                } else {
                    val += bias[n];
                    if (n < 512) of0[(size_t)m * D_MODEL + n] = val;
                    else         of1[(size_t)m * D_MODEL + (n - 512)] = val;
                }
            }
        }
    }
}

// ---------------------------------------------------------------------------
// Flash attention: one block per (128 queries, b, h); wave = 16 query rows.
// scores = Q K^T / 8 + mask ; online softmax ; ctx = P V (V pre-transposed).
// ---------------------------------------------------------------------------
__global__ __launch_bounds__(256) void flash_attn(
    const unsigned short* __restrict__ Qb,
    const unsigned short* __restrict__ Kb,
    const unsigned short* __restrict__ Vt,
    const float* __restrict__ mask,
    unsigned short* __restrict__ ctx)
{
    const int bh    = blockIdx.y;
    const int b     = bh >> 3;
    const int h     = bh & 7;
    const int wave  = threadIdx.x >> 5;
    const int lane  = threadIdx.x & 31;
    const int lhalf = lane >> 4;
    const int l15   = lane & 15;
    const int q0    = blockIdx.x * 128 + wave * 16;

    __shared__ unsigned short Plds[8][16 * 72];   // per-wave P staging (pad->144B rows)

    // Q fragments (16 rows x 64 channels -> 2 A-frags), loaded once
    FragB16 qf[2];
    {
        const unsigned short* qp =
            Qb + ((size_t)(b * SEQ + q0 + l15)) * D_MODEL + h * DK + lhalf * 8;
#pragma unroll
        for (int f = 0; f < 2; ++f) {
            qf[f].q[0] = *reinterpret_cast<const uint4*>(qp + f * 32);
            qf[f].q[1] = *reinterpret_cast<const uint4*>(qp + f * 32 + 16);
        }
    }

    v8f oacc[4];
#pragma unroll
    for (int ct = 0; ct < 4; ++ct)
        oacc[ct] = (v8f){0.f, 0.f, 0.f, 0.f, 0.f, 0.f, 0.f, 0.f};
    float mrun[8], lrun[8];
#pragma unroll
    for (int r = 0; r < 8; ++r) { mrun[r] = -1e30f; lrun[r] = 0.f; }

    for (int n0 = 0; n0 < SEQ; n0 += 64) {
        // ---- scores: 4 key tiles x (2 WMMA over dk=64) ----
        v8f s[4];
#pragma unroll
        for (int nt = 0; nt < 4; ++nt)
            s[nt] = (v8f){0.f, 0.f, 0.f, 0.f, 0.f, 0.f, 0.f, 0.f};
#pragma unroll
        for (int nt = 0; nt < 4; ++nt) {
            const unsigned short* kp =
                Kb + ((size_t)(b * SEQ + n0 + nt * 16 + l15)) * D_MODEL + h * DK + lhalf * 8;
            FragB16 kf0, kf1;
            kf0.q[0] = *reinterpret_cast<const uint4*>(kp);
            kf0.q[1] = *reinterpret_cast<const uint4*>(kp + 16);
            kf1.q[0] = *reinterpret_cast<const uint4*>(kp + 32);
            kf1.q[1] = *reinterpret_cast<const uint4*>(kp + 48);
            s[nt] = __builtin_amdgcn_wmma_f32_16x16x32_bf16(
                false, qf[0].v, false, kf0.v, (short)0, s[nt], false, false);
            s[nt] = __builtin_amdgcn_wmma_f32_16x16x32_bf16(
                false, qf[1].v, false, kf1.v, (short)0, s[nt], false, false);
        }

        // ---- scale + mask + online softmax (per-row stats live per-VGPR) ----
        float mval[4];
#pragma unroll
        for (int nt = 0; nt < 4; ++nt)
            mval[nt] = mask[(size_t)b * SEQ + n0 + nt * 16 + l15];

#pragma unroll
        for (int r = 0; r < 8; ++r) {
            float rmax = -1e30f;
#pragma unroll
            for (int nt = 0; nt < 4; ++nt) {
                float x = s[nt][r] * 0.125f + mval[nt];
                s[nt][r] = x;
                rmax = fmaxf(rmax, x);
            }
#pragma unroll
            for (int off = 8; off >= 1; off >>= 1)
                rmax = fmaxf(rmax, __shfl_xor(rmax, off, 32));
            float mnew = fmaxf(mrun[r], rmax);
            float corr = __expf(mrun[r] - mnew);
            float rsum = 0.f;
#pragma unroll
            for (int nt = 0; nt < 4; ++nt) {
                float p = __expf(s[nt][r] - mnew);
                s[nt][r] = p;
                rsum += p;
            }
#pragma unroll
            for (int off = 8; off >= 1; off >>= 1)
                rsum += __shfl_xor(rsum, off, 32);
            lrun[r] = lrun[r] * corr + rsum;
            mrun[r] = mnew;
#pragma unroll
            for (int ct = 0; ct < 4; ++ct)
                oacc[ct][r] *= corr;
        }

        // ---- P (C-layout) -> LDS -> A-layout fragments ----
        unsigned short* pl = &Plds[wave][0];
#pragma unroll
        for (int r = 0; r < 8; ++r) {
            int prow = lhalf * 8 + r;
#pragma unroll
            for (int nt = 0; nt < 4; ++nt)
                pl[prow * 72 + nt * 16 + l15] = f32_to_bf16(s[nt][r]);
        }
        __syncthreads();

        FragB16 pf[2];
#pragma unroll
        for (int f = 0; f < 2; ++f) {
            const unsigned short* pp = pl + l15 * 72 + f * 32 + lhalf * 8;
            pf[f].q[0] = *reinterpret_cast<const uint4*>(pp);
            pf[f].q[1] = *reinterpret_cast<const uint4*>(pp + 16);
        }

        // ---- ctx += P @ V ; Vt[b,h,ch,s] gives 16B B-fragment chunks (k = 8h.., 16+8h..) ----
#pragma unroll
        for (int ct = 0; ct < 4; ++ct) {
            const unsigned short* vp =
                Vt + (((size_t)((b * NHEADS + h) * DK + ct * 16 + l15)) << 11)
                   + n0 + lhalf * 8;
            FragB16 vf0, vf1;
            vf0.q[0] = *reinterpret_cast<const uint4*>(vp);
            vf0.q[1] = *reinterpret_cast<const uint4*>(vp + 16);
            vf1.q[0] = *reinterpret_cast<const uint4*>(vp + 32);
            vf1.q[1] = *reinterpret_cast<const uint4*>(vp + 48);
            oacc[ct] = __builtin_amdgcn_wmma_f32_16x16x32_bf16(
                false, pf[0].v, false, vf0.v, (short)0, oacc[ct], false, false);
            oacc[ct] = __builtin_amdgcn_wmma_f32_16x16x32_bf16(
                false, pf[1].v, false, vf1.v, (short)0, oacc[ct], false, false);
        }
    }

    // ---- normalize + write ctx bf16 [B,S,H*dk] ----
#pragma unroll
    for (int ct = 0; ct < 4; ++ct) {
#pragma unroll
        for (int r = 0; r < 8; ++r) {
            int m   = q0 + lhalf * 8 + r;
            int chn = ct * 16 + l15;
            float val = oacc[ct][r] / lrun[r];
            ctx[((size_t)(b * SEQ + m)) * D_MODEL + h * DK + chn] = f32_to_bf16(val);
        }
    }
}

// ---------------------------------------------------------------------------
// g = relu(o) * sigmoid(gp); LayerNorm over D=512 -> bf16
// ---------------------------------------------------------------------------
__global__ __launch_bounds__(256) void gate_ln(
    const float* __restrict__ o, const float* __restrict__ gp,
    const float* __restrict__ gamma, const float* __restrict__ beta,
    unsigned short* __restrict__ ln)
{
    const int row = blockIdx.x;
    const int tid = threadIdx.x;
    const size_t base = (size_t)row * D_MODEL;
    __shared__ float rs[2][8];

    float ov0 = o[base + tid],       gv0 = gp[base + tid];
    float ov1 = o[base + tid + 256], gv1 = gp[base + tid + 256];
    float g0 = fmaxf(ov0, 0.f) * (1.f / (1.f + __expf(-gv0)));
    float g1 = fmaxf(ov1, 0.f) * (1.f / (1.f + __expf(-gv1)));

    float sum = g0 + g1, sq = g0 * g0 + g1 * g1;
#pragma unroll
    for (int off = 16; off >= 1; off >>= 1) {
        sum += __shfl_xor(sum, off, 32);
        sq  += __shfl_xor(sq,  off, 32);
    }
    if ((tid & 31) == 0) { rs[0][tid >> 5] = sum; rs[1][tid >> 5] = sq; }
    __syncthreads();
    float tsum = 0.f, tsq = 0.f;
#pragma unroll
    for (int i = 0; i < 8; ++i) { tsum += rs[0][i]; tsq += rs[1][i]; }

    float mean = tsum * (1.f / D_MODEL);
    float var  = tsq * (1.f / D_MODEL) - mean * mean;
    float rstd = rsqrtf(var + 1e-6f);
    ln[base + tid]       = f32_to_bf16((g0 - mean) * rstd * gamma[tid] + beta[tid]);
    ln[base + tid + 256] = f32_to_bf16((g1 - mean) * rstd * gamma[tid + 256] + beta[tid + 256]);
}

// ---------------------------------------------------------------------------
// ret = mu + exp(0.5*lv)*noise ; priorkl = -0.5 * sum(1+lv-mu^2-exp(lv))
// ---------------------------------------------------------------------------
__global__ __launch_bounds__(256) void vib_out(
    const float* __restrict__ mu, const float* __restrict__ lv,
    const float* __restrict__ noise,
    float* __restrict__ ret, float* __restrict__ kl)
{
    const int row = blockIdx.x;
    const int tid = threadIdx.x;
    const size_t base = (size_t)row * D_MODEL;
    __shared__ float rs[8];

    float acc = 0.f;
#pragma unroll
    for (int j = 0; j < 2; ++j) {
        int idx = tid + j * 256;
        float m = mu[base + idx], l = lv[base + idx], nz = noise[base + idx];
        float el = __expf(l);
        ret[base + idx] = m + __expf(0.5f * l) * nz;
        acc += 1.f + l - m * m - el;
    }
#pragma unroll
    for (int off = 16; off >= 1; off >>= 1) acc += __shfl_xor(acc, off, 32);
    if ((tid & 31) == 0) rs[tid >> 5] = acc;
    __syncthreads();
    if (tid == 0) {
        float t = 0.f;
#pragma unroll
        for (int i = 0; i < 8; ++i) t += rs[i];
        kl[row] = -0.5f * t;
    }
}

// ---------------------------------------------------------------------------
extern "C" void kernel_launch(void* const* d_in, const int* in_sizes, int n_in,
                              void* d_out, int out_size, void* d_ws, size_t ws_size,
                              hipStream_t stream) {
    (void)in_sizes; (void)n_in; (void)out_size; (void)ws_size;

    const float* input = (const float*)d_in[0];
    const float* mask  = (const float*)d_in[1];
    const float* noise = (const float*)d_in[2];
    const float* Wq    = (const float*)d_in[3];
    const float* Wk    = (const float*)d_in[4];
    const float* Wv    = (const float*)d_in[5];
    const float* Wo    = (const float*)d_in[6];
    const float* Wg    = (const float*)d_in[7];
    const float* Wmu   = (const float*)d_in[8];
    const float* bmu   = (const float*)d_in[9];
    const float* Wlv   = (const float*)d_in[10];
    const float* blv   = (const float*)d_in[11];
    const float* gamma = (const float*)d_in[12];
    const float* beta  = (const float*)d_in[13];

    char* w = (char*)d_ws;
    size_t off = 0;
    auto take = [&](size_t bytes) -> char* {
        char* p = w + off;
        off += bytes;
        off = (off + 255) & ~(size_t)255;
        return p;
    };

    const size_t HB = (size_t)MTOT * D_MODEL * 2;   // bf16 activation buffer
    const size_t FB = (size_t)MTOT * D_MODEL * 4;   // f32 activation buffer

    unsigned short* Xb    = (unsigned short*)take(HB);
    unsigned short* Wqkv  = (unsigned short*)take((size_t)1536 * 512 * 2);
    unsigned short* Wog   = (unsigned short*)take((size_t)1024 * 512 * 2);
    unsigned short* Wmlv  = (unsigned short*)take((size_t)1024 * 512 * 2);
    float*          bcat  = (float*)take(1024 * 4);
    unsigned short* Qb    = (unsigned short*)take(HB);
    unsigned short* Kb    = (unsigned short*)take(HB);
    unsigned short* Vt    = (unsigned short*)take(HB);
    unsigned short* Ctx   = (unsigned short*)take(HB);
    float*          Obuf  = (float*)take(FB);
    float*          Gpbuf = (float*)take(FB);
    unsigned short* Lnb   = (unsigned short*)take(HB);
    float*          Mu    = (float*)take(FB);
    float*          Lv    = (float*)take(FB);

    // 1) convert activations + weights to bf16 (weights concatenated)
    {
        int n4 = (MTOT * D_MODEL) / 4;
        cvt_f32_bf16_v4<<<(n4 + 255) / 256, 256, 0, stream>>>(input, Xb, n4);
        n4 = (512 * 512) / 4;
        cvt_f32_bf16_v4<<<(n4 + 255) / 256, 256, 0, stream>>>(Wq, Wqkv, n4);
        cvt_f32_bf16_v4<<<(n4 + 255) / 256, 256, 0, stream>>>(Wk, Wqkv + 262144, n4);
        cvt_f32_bf16_v4<<<(n4 + 255) / 256, 256, 0, stream>>>(Wv, Wqkv + 524288, n4);
        cvt_f32_bf16_v4<<<(n4 + 255) / 256, 256, 0, stream>>>(Wo, Wog, n4);
        cvt_f32_bf16_v4<<<(n4 + 255) / 256, 256, 0, stream>>>(Wg, Wog + 262144, n4);
        cvt_f32_bf16_v4<<<(n4 + 255) / 256, 256, 0, stream>>>(Wmu, Wmlv, n4);
        cvt_f32_bf16_v4<<<(n4 + 255) / 256, 256, 0, stream>>>(Wlv, Wmlv + 262144, n4);
        copy_f32<<<2, 256, 0, stream>>>(bmu, bcat, 512);
        copy_f32<<<2, 256, 0, stream>>>(blv, bcat + 512, 512);
    }

    // 2) fused QKV projection (N=1536), V written transposed [B,H,dk,S]
    gemm_bf16<<<dim3(MTOT / 128, 1536 / 128), 256, 0, stream>>>(
        Xb, Wqkv, 0, Qb, Kb, Vt, nullptr, nullptr, nullptr);

    // 3) flash attention
    flash_attn<<<dim3(SEQ / 128, BATCH * NHEADS), 256, 0, stream>>>(
        Qb, Kb, Vt, mask, Ctx);

    // 4) fused O|G projection (N=1024)
    gemm_bf16<<<dim3(MTOT / 128, 1024 / 128), 256, 0, stream>>>(
        Ctx, Wog, 1, nullptr, nullptr, nullptr, Obuf, Gpbuf, nullptr);

    // 5) gate + LayerNorm
    gate_ln<<<MTOT, 256, 0, stream>>>(Obuf, Gpbuf, gamma, beta, Lnb);

    // 6) fused mu|logvar projection (N=1024) with bias
    gemm_bf16<<<dim3(MTOT / 128, 1024 / 128), 256, 0, stream>>>(
        Lnb, Wmlv, 2, nullptr, nullptr, nullptr, Mu, Lv, bcat);

    // 7) reparameterize + prior KL
    vib_out<<<MTOT, 256, 0, stream>>>(
        Mu, Lv, noise, (float*)d_out, (float*)d_out + (size_t)MTOT * D_MODEL);
}
